// _GATHead_63608465654521
// MI455X (gfx1250) — matrix operand order repaired and checked by reading, not compiled, not run on hardware
//
#include <hip/hip_runtime.h>
#include <hip/hip_bf16.h>
#include <math.h>

// ---- CDNA5 WMMA types ----
typedef __attribute__((ext_vector_type(16))) _Float16     v16h;
typedef __attribute__((ext_vector_type(8)))  float        v8f;
typedef __attribute__((ext_vector_type(4)))  unsigned int v4u;

union H16x16 { v4u u[2]; v16h h; };

// Problem constants (from reference setup_inputs)
#define BB   8
#define NN   2048
#define IND  256
#define OUTD 128
#define ROWS (BB * NN)          // 16384

// ---------------------------------------------------------------------------
// K0: Wt[o][k] = (f16) W[o][k]   (W is OUTD x IND row-major)
// ---------------------------------------------------------------------------
__global__ void gat_k0_wt(const float* __restrict__ W, _Float16* __restrict__ Wt) {
    int i = blockIdx.x * 256 + threadIdx.x;          // over OUTD*IND = 32768
    if (i >= OUTD * IND) return;
    int o = i / IND;
    int k = i - o * IND;
    Wt[(size_t)o * IND + k] = (_Float16)W[i];
}

// ---------------------------------------------------------------------------
// K1: h = x @ W^T  via v_wmma_f32_16x16x32_f16.
// The 16x256 f32 x row-block (16 KB) is staged into LDS ONCE per block via
// the CDNA5 async global->LDS copy (global_load_async_to_lds_b128, tracked by
// ASYNCcnt), removing the 8x per-wave A-fragment read redundancy. Note the
// ISA adds INST_OFFSET to BOTH global and LDS addresses, so a contiguous
// block copy only needs per-thread base addresses plus offset: immediates.
// ---------------------------------------------------------------------------
__global__ void gat_k1_gemm(const float* __restrict__ x,
                            const _Float16* __restrict__ Wt,
                            float* __restrict__ h32,
                            _Float16* __restrict__ h16) {
    __shared__ __align__(16) float XS[16 * IND];     // 16 KB staged x tile

    const int t    = threadIdx.x;
    const int lane = t & 31;
    const int w    = t >> 5;              // 0..7  (o-tile index)
    const int row0 = blockIdx.x * 16;
    const int o0   = w * 16;

    // ---- async stage: 256 threads x 4 x b128 = 16 KB ----
    {
        const unsigned long long g0 =
            (unsigned long long)(const void*)(x + (size_t)row0 * IND) + (unsigned)(t * 16);
        const unsigned l0 = (unsigned)(size_t)(&XS[0]) + (unsigned)(t * 16);
        asm volatile("global_load_async_to_lds_b128 %0, %1, off"
                     :: "v"(l0), "v"(g0) : "memory");
        asm volatile("global_load_async_to_lds_b128 %0, %1, off offset:4096"
                     :: "v"(l0), "v"(g0) : "memory");
        asm volatile("global_load_async_to_lds_b128 %0, %1, off offset:8192"
                     :: "v"(l0), "v"(g0) : "memory");
        asm volatile("global_load_async_to_lds_b128 %0, %1, off offset:12288"
                     :: "v"(l0), "v"(g0) : "memory");
        asm volatile("s_wait_asynccnt 0x0" ::: "memory");
    }
    __syncthreads();

    const int m  = lane & 15;             // A-matrix row within tile
    const int kb = (lane < 16) ? 0 : 8;   // A-matrix K sub-offset per lane half

    v8f acc = {0.f,0.f,0.f,0.f,0.f,0.f,0.f,0.f};

    #pragma unroll
    for (int ks = 0; ks < IND; ks += 32) {
        // A fragment from LDS: XS[m][ks+kb .. +7] and XS[m][ks+16+kb .. +7]
        const float* ap = &XS[m * IND + ks + kb];
        v16h a;
        #pragma unroll
        for (int j = 0; j < 8; ++j) {
            a[j]     = (_Float16)ap[j];
            a[j + 8] = (_Float16)ap[16 + j];
        }
        // B fragment: Wt[o0+m][ks+kb .. +7] and Wt[o0+m][ks+16+kb .. +7]
        const _Float16* bp = &Wt[(size_t)(o0 + m) * IND + ks + kb];
        H16x16 ub;
        ub.u[0] = *(const v4u*)(bp);
        ub.u[1] = *(const v4u*)(bp + 16);

        acc = __builtin_amdgcn_wmma_f32_16x16x32_f16(
            false, a, false, ub.h, (short)0, acc, false, false);
    }

    // C/D layout: VGPR r -> M = r + (lane<16 ? 0 : 8), N = lane & 15
    const int rbase = (lane < 16) ? 0 : 8;
    #pragma unroll
    for (int r = 0; r < 8; ++r) {
        const int    row = row0 + r + rbase;
        const size_t idx = (size_t)row * OUTD + o0 + m;
        const float  v   = acc[r];
        h32[idx] = v;
        h16[idx] = (_Float16)v;
    }
}

// ---------------------------------------------------------------------------
// K2: src[row] = h[row,:] . a_src ; dst[row] = h[row,:] . a_dst
// One wave per row (8 rows per 256-thread block).
// ---------------------------------------------------------------------------
__global__ void gat_k2_srcdst(const float* __restrict__ h32,
                              const float* __restrict__ a_src,
                              const float* __restrict__ a_dst,
                              float* __restrict__ srcv,
                              float* __restrict__ dstv) {
    const int lane = threadIdx.x & 31;
    const int w    = threadIdx.x >> 5;
    const int row  = blockIdx.x * 8 + w;
    const float* hp = &h32[(size_t)row * OUTD];

    float s = 0.f, d = 0.f;
    #pragma unroll
    for (int c = 0; c < 4; ++c) {
        const int   o  = lane + 32 * c;
        const float hv = hp[o];
        s += hv * a_src[o];
        d += hv * a_dst[o];
    }
    #pragma unroll
    for (int off = 16; off > 0; off >>= 1) {
        s += __shfl_xor(s, off, 32);
        d += __shfl_xor(d, off, 32);
    }
    if (lane == 0) { srcv[row] = s; dstv[row] = d; }
}

// ---------------------------------------------------------------------------
// K3: Dmax[b] = max over unmasked j of dst[b,j]   (sentinel -1e38 if none)
// ---------------------------------------------------------------------------
__global__ void gat_k3_dmax(const float* __restrict__ dstv,
                            const int* __restrict__ mask,
                            float* __restrict__ Dmax) {
    __shared__ float red[256];
    const int b = blockIdx.x;
    const int t = threadIdx.x;
    float mx = -1e38f;
    for (int j = t; j < NN; j += 256) {
        if (mask[(size_t)b * NN + j] != 0)
            mx = fmaxf(mx, dstv[(size_t)b * NN + j]);
    }
    red[t] = mx;
    __syncthreads();
    #pragma unroll
    for (int s = 128; s > 0; s >>= 1) {
        if (t < s) red[t] = fmaxf(red[t], red[t + s]);
        __syncthreads();
    }
    if (t == 0) Dmax[b] = red[0];
}

// ---------------------------------------------------------------------------
// K4: fused masked softmax + (alpha @ h), 32-row tile per block.
// Exact row max M_i = lrelu(src_i + Dmax_b) (leaky_relu is monotone) -> one
// pass, no online rescale. Per iteration: a 32x32 P tile (4 exp/thread) is
// written into a DOUBLE-BUFFERED LDS tile (single barrier/iter: writes to
// buffer jt&1 at iter jt+2 are fenced from iter-jt reads by barrier jt+1);
// each of 8 waves runs TWO WMMAs sharing one B fragment (B reuse). Row sums
// Z live in registers (thread t's P elements sit in rows w,w+8,w+16,w+24);
// wave-local shfl reduce -> deterministic, no atomics.
// ---------------------------------------------------------------------------
__global__ void gat_k4_attn(const float* __restrict__ srcv,
                            const float* __restrict__ dstv,
                            const int* __restrict__ mask,
                            const float* __restrict__ Dmax,
                            const _Float16* __restrict__ h16,
                            float* __restrict__ out) {
    __shared__ __align__(16) _Float16 P[2][32 * 32];   // double-buffered, 4 KB
    __shared__ float sSrc[32];
    __shared__ float sM[32];
    __shared__ float sZ[32];

    const int bidx = blockIdx.x;          // 512 blocks
    const int b    = bidx >> 6;           // 64 row-tiles (of 32 rows) per batch
    const int it   = bidx & 63;
    const int i0   = it * 32;

    const int t    = threadIdx.x;
    const int lane = t & 31;
    const int w    = t >> 5;              // wave = o-tile index AND row-partial index
    const int o0   = w * 16;

    if (t < 32) {
        const float s  = srcv[(size_t)b * NN + i0 + t];
        const float dm = Dmax[b];
        float e = s + dm;
        e = (e > 0.f) ? e : 0.2f * e;
        sSrc[t] = s;
        sM[t]   = (dm < -1e37f) ? 0.f : e;   // all-masked batch guard
    }
    __syncthreads();

    float srcR[4], mR[4];
    #pragma unroll
    for (int q = 0; q < 4; ++q) { srcR[q] = sSrc[w + 8 * q]; mR[q] = sM[w + 8 * q]; }

    const int am = lane & 15;             // A-frag row
    const int kb = (lane < 16) ? 0 : 8;   // A-frag K sub-offset

    float z[4] = {0.f, 0.f, 0.f, 0.f};
    v8f acc0 = {0.f,0.f,0.f,0.f,0.f,0.f,0.f,0.f};   // rows i0+0..15
    v8f acc1 = {0.f,0.f,0.f,0.f,0.f,0.f,0.f,0.f};   // rows i0+16..31

    const _Float16* bbase = &h16[(size_t)b * NN * OUTD + o0];

    for (int jt = 0; jt < NN / 32; ++jt) {
        const int   j  = jt * 32 + lane;  // P column == K row of B tile
        const float d  = dstv[(size_t)b * NN + j];
        const int   mk = mask[(size_t)b * NN + j];

        _Float16* Pb = &P[jt & 1][0];
        float p[4];
        #pragma unroll
        for (int q = 0; q < 4; ++q) {
            float xv = srcR[q] + d;
            xv = (xv > 0.f) ? xv : 0.2f * xv;
            p[q] = mk ? __expf(xv - mR[q]) : 0.f;
            z[q] += p[q];
        }
        #pragma unroll
        for (int q = 0; q < 4; ++q)
            Pb[(w + 8 * q) * 32 + lane] = (_Float16)p[q];

        __syncthreads();                  // single barrier (double buffering)

        H16x16 ua0, ua1, ub;
        ua0.u[0] = *(const v4u*)(&Pb[am * 32 + kb]);
        ua0.u[1] = *(const v4u*)(&Pb[am * 32 + 16 + kb]);
        ua1.u[0] = *(const v4u*)(&Pb[(16 + am) * 32 + kb]);
        ua1.u[1] = *(const v4u*)(&Pb[(16 + am) * 32 + 16 + kb]);

        // B fragment: h16[b*NN + jt*32 + kb+s / 16+kb+s][o0 + am]
        const _Float16* bp = bbase + (size_t)j * OUTD;
        #pragma unroll
        for (int s = 0; s < 8; ++s) {
            ub.h[s]     = bbase[(size_t)(jt * 32 + kb + s) * OUTD + am];
            ub.h[s + 8] = bbase[(size_t)(jt * 32 + 16 + kb + s) * OUTD + am];
        }
        __builtin_prefetch(bp + 32 * OUTD, 0, 1);   // next jt's B tile

        acc0 = __builtin_amdgcn_wmma_f32_16x16x32_f16(
            false, ua0.h, false, ub.h, (short)0, acc0, false, false);
        acc1 = __builtin_amdgcn_wmma_f32_16x16x32_f16(
            false, ua1.h, false, ub.h, (short)0, acc1, false, false);
    }

    // deterministic row-sum reduction (no atomics)
    #pragma unroll
    for (int off = 16; off > 0; off >>= 1) {
        #pragma unroll
        for (int q = 0; q < 4; ++q) z[q] += __shfl_xor(z[q], off, 32);
    }
    if (lane == 0) {
        sZ[w] = z[0]; sZ[w + 8] = z[1]; sZ[w + 16] = z[2]; sZ[w + 24] = z[3];
    }
    __syncthreads();

    const int rbase = (lane < 16) ? 0 : 8;
    #pragma unroll
    for (int r = 0; r < 8; ++r) {
        const int   row0 = r + rbase;          // tile-0 row
        const int   row1 = 16 + row0;          // tile-1 row
        const float z0   = sZ[row0];
        const float z1   = sZ[row1];
        const float v0   = (z0 > 0.f) ? acc0[r] / z0 : 0.f;  // all-masked -> 0
        const float v1   = (z1 > 0.f) ? acc1[r] / z1 : 0.f;
        out[((size_t)b * NN + i0 + row0) * OUTD + o0 + am] = v0;
        out[((size_t)b * NN + i0 + row1) * OUTD + o0 + am] = v1;
    }
}

// ---------------------------------------------------------------------------
extern "C" void kernel_launch(void* const* d_in, const int* in_sizes, int n_in,
                              void* d_out, int out_size, void* d_ws, size_t ws_size,
                              hipStream_t stream) {
    const float* x     = (const float*)d_in[0];   // (8,2048,256)
    const int*   mask  = (const int*)  d_in[1];   // (8,2048)
    const float* W     = (const float*)d_in[2];   // (128,256)
    const float* a_src = (const float*)d_in[3];   // (1,128)
    const float* a_dst = (const float*)d_in[4];   // (1,128)
    float*       out   = (float*)d_out;           // (8,2048,128)

    // workspace layout (~12.2 MB)
    char* ws = (char*)d_ws;
    float*     h32  = (float*)ws;                                   // 8 MB
    _Float16*  h16  = (_Float16*)(ws + (size_t)ROWS * OUTD * 4);    // 4 MB
    _Float16*  Wt   = (_Float16*)(ws + (size_t)ROWS * OUTD * 6);    // 64 KB
    float*     srcv = (float*)(ws + (size_t)ROWS * OUTD * 6 + IND * OUTD * 2);
    float*     dstv = srcv + ROWS;
    float*     Dmax = dstv + ROWS;

    gat_k0_wt    <<<(OUTD * IND + 255) / 256, 256, 0, stream>>>(W, Wt);
    gat_k1_gemm  <<<ROWS / 16,  256, 0, stream>>>(x, Wt, h32, h16);
    gat_k2_srcdst<<<ROWS / 8,   256, 0, stream>>>(h32, a_src, a_dst, srcv, dstv);
    gat_k3_dmax  <<<BB,         256, 0, stream>>>(dstv, mask, Dmax);
    gat_k4_attn  <<<BB * (NN / 32), 256, 0, stream>>>(srcv, dstv, mask, Dmax, h16, out);
}
